// Attention_30571577213493
// MI455X (gfx1250) — matrix operand order repaired and checked by reading, not compiled, not run on hardware
//
#include <hip/hip_runtime.h>

// ---------------- types ----------------
typedef __bf16 bf16_t;
typedef __attribute__((ext_vector_type(16))) __bf16 v16bf;
typedef __attribute__((ext_vector_type(8)))  __bf16 v8bf;
typedef __attribute__((ext_vector_type(4)))  __bf16 v4bf;
typedef __attribute__((ext_vector_type(8)))  float  v8f;
typedef __attribute__((ext_vector_type(4)))  unsigned int v4u;
typedef __attribute__((ext_vector_type(8)))  int v8i;
typedef __attribute__((ext_vector_type(4)))  int v4i;

#define WMMA_BF16(a, b, c) \
  __builtin_amdgcn_wmma_f32_16x16x32_bf16(false, (a), false, (b), (short)0, (c), false, false)

#define IGNORE_VAL (-100000.0f)

#if defined(__has_builtin)
#if __has_builtin(__builtin_amdgcn_tensor_load_to_lds) && __has_builtin(__builtin_amdgcn_s_wait_tensorcnt)
#define USE_TDM 1
#endif
#endif
#ifndef USE_TDM
#define USE_TDM 0
#endif

__device__ __forceinline__ v16bf ld16(const bf16_t* p) {
  return *(const v16bf*)p;  // 32B contiguous (B operand)
}
// A-operand: 16-bit A layout = two contiguous 8-elt chunks at K-offsets +0 and +16
__device__ __forceinline__ v16bf ld8x2(const bf16_t* p) {
  v8bf lo = *(const v8bf*)p;
  v8bf hi = *(const v8bf*)(p + 16);
  return __builtin_shufflevector(lo, hi, 0,1,2,3,4,5,6,7,8,9,10,11,12,13,14,15);
}

#if USE_TDM
// low 32 bits of a generic pointer to LDS == LDS byte address
__device__ __forceinline__ unsigned lds_addr32(const void* p) {
  return (unsigned)(uintptr_t)p;
}
// 2D tile DMA: tile_k contiguous bf16 along dim0, tile_rows lines, row stride in elements.
// D# per CDNA5 ISA ch.8: group0 {count=1, lds_addr, global_addr, type=2},
// group1 {data_size=1(2B), tensor_dim0=tile_k, tensor_dim1=tile_rows, tile dims, stride0}.
__device__ __forceinline__ void tdm_load_2d(unsigned lds_addr, const void* gsrc,
                                            unsigned tile_k, unsigned tile_rows,
                                            unsigned row_stride_elts) {
  unsigned long long ga = (unsigned long long)(uintptr_t)gsrc;
  v4u g0 = { 1u,                                        // count=1 (valid user desc)
             lds_addr,                                  // [63:32] LDS byte address
             (unsigned)ga,                              // global_addr[31:0]
             (unsigned)((ga >> 32) & 0x01ffffffu) | 0x80000000u };  // addr[56:32] | type=2
  v8i g1 = { (int)0x10000u,                             // data_size=1 (2 bytes)
             (int)((tile_k & 0xffffu) << 16),           // tensor_dim0[15:0]
             (int)((tile_k >> 16) | ((tile_rows & 0xffffu) << 16)),  // dim0 hi | dim1 lo
             (int)((tile_rows >> 16) | (tile_k << 16)), // dim1 hi | tile_dim0
             (int)(tile_rows & 0xffffu),                // tile_dim1 (tile_dim2=0)
             (int)row_stride_elts,                      // tensor_dim0_stride[31:0]
             0, 0 };
  v4i gz = { 0, 0, 0, 0 };
#if __clang_major__ >= 23
  v8i gz8 = { 0, 0, 0, 0, 0, 0, 0, 0 };
  __builtin_amdgcn_tensor_load_to_lds(g0, g1, gz, gz, gz8, 0);
#else
  __builtin_amdgcn_tensor_load_to_lds(g0, g1, gz, gz, 0);
#endif
}
#endif  // USE_TDM

// ---------------- pre-pass: fp32 -> bf16 (vectorized x4) ----------------
__global__ __launch_bounds__(256) void cvt_bf16(const float4* __restrict__ s,
                                                v4bf* __restrict__ d) {
  size_t i = (size_t)blockIdx.x * blockDim.x + threadIdx.x;
  float4 f = s[i];
  v4bf o;
  o[0] = (bf16_t)f.x; o[1] = (bf16_t)f.y; o[2] = (bf16_t)f.z; o[3] = (bf16_t)f.w;
  d[i] = o;
}

// ---------------- pre-pass: weight transpose to Bt[col][k] bf16 ----------------
template <bool QKV>
__global__ __launch_bounds__(256) void wtrans(const float* __restrict__ src,
                                              bf16_t* __restrict__ dst) {
  int gid = blockIdx.x * 256 + threadIdx.x;   // 0 .. 1048575
  int r = gid >> 10, c = gid & 1023;
  float v;
  if (QKV) v = src[((r >> 6) << 16) + (c << 6) + (r & 63)];
  else     v = src[c * 1024 + r];
  dst[gid] = (bf16_t)v;
}

// ---------------- bf16 GEMM: C[8192x1024] = A[8192x1024] @ Bt^T + bias ----------------
// MODE bit0: fp32 out; bit1: bf16 out; bit2: bf16 transposed [b][n][h][s]
// TDM double-buffered LDS staging: per K-step wave0 DMAs A(128x32) and B(128x32) tiles.
template <int MODE>
__global__ __launch_bounds__(256) void gemm_bf16(const bf16_t* __restrict__ A,
                                                 const bf16_t* __restrict__ Bt,
                                                 const float*  __restrict__ bias,
                                                 float*  __restrict__ outF,
                                                 bf16_t* __restrict__ outBf,
                                                 bf16_t* __restrict__ outBfT) {
  const int lane = threadIdx.x & 31;
  const int wave = threadIdx.x >> 5;
  const int l16  = lane & 15;
  const int lh   = lane >> 4;
  const int rowB = blockIdx.x * 128;
  const int colB = blockIdx.y * 128;
  const int wr = (wave & 3) * 32;   // wave row offset inside block tile
  const int wc = (wave >> 2) * 64;  // wave col offset inside block tile

  v8f acc[2][4] = {};

#if USE_TDM
  __shared__ __align__(128) bf16_t shA[2][128 * 32];
  __shared__ __align__(128) bf16_t shB[2][128 * 32];
  const bf16_t* gA = A  + (size_t)rowB * 1024;
  const bf16_t* gB = Bt + (size_t)colB * 1024;

  if (wave == 0) {
    tdm_load_2d(lds_addr32(&shA[0][0]), gA, 32, 128, 1024);
    tdm_load_2d(lds_addr32(&shB[0][0]), gB, 32, 128, 1024);
  }
  for (int it = 0; it < 32; ++it) {
    const int buf = it & 1;
    if (wave == 0) __builtin_amdgcn_s_wait_tensorcnt(0);   // buf tiles landed in LDS
    __syncthreads();  // also fences all reads of buf^1 from iter it-1 before overwrite
    if (wave == 0 && it + 1 < 32) {
      tdm_load_2d(lds_addr32(&shA[buf ^ 1][0]), gA + (size_t)(it + 1) * 32, 32, 128, 1024);
      tdm_load_2d(lds_addr32(&shB[buf ^ 1][0]), gB + (size_t)(it + 1) * 32, 32, 128, 1024);
    }
    const bf16_t* sa = &shA[buf][0];
    const bf16_t* sb = &shB[buf][0];
    v16bf a0 = ld16(sa + (wr + l16) * 32 + lh * 16);
    v16bf a1 = ld16(sa + (wr + 16 + l16) * 32 + lh * 16);
    v16bf b0 = ld16(sb + (wc + l16) * 32 + lh * 16);
    v16bf b1 = ld16(sb + (wc + 16 + l16) * 32 + lh * 16);
    v16bf b2 = ld16(sb + (wc + 32 + l16) * 32 + lh * 16);
    v16bf b3 = ld16(sb + (wc + 48 + l16) * 32 + lh * 16);
    acc[0][0] = WMMA_BF16(a0, b0, acc[0][0]);
    acc[0][1] = WMMA_BF16(a0, b1, acc[0][1]);
    acc[0][2] = WMMA_BF16(a0, b2, acc[0][2]);
    acc[0][3] = WMMA_BF16(a0, b3, acc[0][3]);
    acc[1][0] = WMMA_BF16(a1, b0, acc[1][0]);
    acc[1][1] = WMMA_BF16(a1, b1, acc[1][1]);
    acc[1][2] = WMMA_BF16(a1, b2, acc[1][2]);
    acc[1][3] = WMMA_BF16(a1, b3, acc[1][3]);
  }
#else
  // Fallback: direct global operand fetch
  const bf16_t* pa0 = A + (size_t)(rowB + wr + l16) * 1024 + lh * 16;
  const bf16_t* pa1 = pa0 + (size_t)16 * 1024;
  const bf16_t* pb[4];
#pragma unroll
  for (int ct = 0; ct < 4; ++ct)
    pb[ct] = Bt + (size_t)(colB + wc + ct * 16 + l16) * 1024 + lh * 16;
  for (int k = 0; k < 1024; k += 32) {
    v16bf a0 = ld16(pa0 + k);
    v16bf a1 = ld16(pa1 + k);
    v16bf b0 = ld16(pb[0] + k);
    v16bf b1 = ld16(pb[1] + k);
    v16bf b2 = ld16(pb[2] + k);
    v16bf b3 = ld16(pb[3] + k);
    acc[0][0] = WMMA_BF16(a0, b0, acc[0][0]);
    acc[0][1] = WMMA_BF16(a0, b1, acc[0][1]);
    acc[0][2] = WMMA_BF16(a0, b2, acc[0][2]);
    acc[0][3] = WMMA_BF16(a0, b3, acc[0][3]);
    acc[1][0] = WMMA_BF16(a1, b0, acc[1][0]);
    acc[1][1] = WMMA_BF16(a1, b1, acc[1][1]);
    acc[1][2] = WMMA_BF16(a1, b2, acc[1][2]);
    acc[1][3] = WMMA_BF16(a1, b3, acc[1][3]);
  }
#endif

  float bv[4];
#pragma unroll
  for (int ct = 0; ct < 4; ++ct) bv[ct] = bias[colB + wc + ct * 16 + l16];

#pragma unroll
  for (int rt = 0; rt < 2; ++rt)
#pragma unroll
    for (int ct = 0; ct < 4; ++ct)
#pragma unroll
      for (int r = 0; r < 8; ++r) {
        float v = acc[rt][ct][r] + bv[ct];
        int row = rowB + wr + rt * 16 + r + lh * 8;  // C/D: M = vgpr + 8*(lane>=16)
        int col = colB + wc + ct * 16 + l16;         //      N = lane%16
        size_t idx = (size_t)row * 1024 + col;
        if constexpr (MODE & 1) outF[idx] = v;
        if constexpr (MODE & 2) outBf[idx] = (bf16_t)v;
        if constexpr (MODE & 4) {
          int bb = row >> 11, s = row & 2047;
          int n = col >> 6, h = col & 63;
          outBfT[(((size_t)bb * 16 + n) * 64 + h) * 2048 + s] = (bf16_t)v;
        }
      }
}

// ---------------- flash attention ----------------
// grid: (SQ/128, N, B); block 256 = 8 waves; each wave owns 16 q rows of one (b, n).
__global__ __launch_bounds__(256) void flash_attn(const bf16_t* __restrict__ Q,
                                                  const bf16_t* __restrict__ K,
                                                  const bf16_t* __restrict__ Vt,
                                                  bf16_t* __restrict__ Z) {
  __shared__ __align__(64) bf16_t pshm[8][512];  // per-wave 16x32 P tile (C->A relayout)
  const int lane = threadIdx.x & 31;
  const int wave = threadIdx.x >> 5;
  const int l16  = lane & 15;
  const int lh   = lane >> 4;
  const int b    = blockIdx.z;
  const int n    = blockIdx.y;
  const int q0   = blockIdx.x * 128 + wave * 16;

  const bf16_t* qrow = Q + (size_t)(b * 2048 + q0 + l16) * 1024 + n * 64;
  v16bf qa0 = ld16(qrow + lh * 16);
  v16bf qa1 = ld16(qrow + 32 + lh * 16);

  v8f z0 = {}, z1 = {}, z2 = {}, z3 = {};
  float mrow[8], lrow[8];
#pragma unroll
  for (int r = 0; r < 8; ++r) { mrow[r] = -3.0e38f; lrow[r] = 0.0f; }

  bf16_t* tile = &pshm[wave][0];
  const size_t kbase = (size_t)b * 2048 * 1024 + n * 64;
  const size_t vbase = (((size_t)b * 16 + n) * 64) * 2048;

  const int kend = q0 + 16;                 // causal: keys kp <= q0+15
  for (int k0 = 0; k0 < kend; k0 += 32) {
    v8f s0 = {}, s1 = {};
    const bf16_t* kr0 = K + kbase + (size_t)(k0 + l16) * 1024 + lh * 16;
    const bf16_t* kr1 = K + kbase + (size_t)(k0 + 16 + l16) * 1024 + lh * 16;
    const bf16_t* vr  = Vt + vbase + (size_t)l16 * 2048 + k0 + lh * 16;
    if (k0 + 32 < kend) {
      __builtin_prefetch(kr0 + (size_t)32 * 1024, 0, 0);  // next K tile
      __builtin_prefetch(vr + 32, 0, 0);                  // next V tile
    }
    s0 = WMMA_BF16(qa0, ld16(kr0), s0);
    s0 = WMMA_BF16(qa1, ld16(kr0 + 32), s0);
    s1 = WMMA_BF16(qa0, ld16(kr1), s1);
    s1 = WMMA_BF16(qa1, ld16(kr1 + 32), s1);

    float fac[8];
#pragma unroll
    for (int r = 0; r < 8; ++r) {
      int qm = q0 + r + lh * 8;
      float v0 = s0[r] * 0.125f;                     // 1/sqrt(64)
      float v1 = s1[r] * 0.125f;
      if (k0 + l16 > qm)      v0 = IGNORE_VAL;       // causal mask (reference IGNORE)
      if (k0 + 16 + l16 > qm) v1 = IGNORE_VAL;
      float bm = fmaxf(v0, v1);
      bm = fmaxf(bm, __shfl_xor(bm, 1));
      bm = fmaxf(bm, __shfl_xor(bm, 2));
      bm = fmaxf(bm, __shfl_xor(bm, 4));
      bm = fmaxf(bm, __shfl_xor(bm, 8));             // reduce within 16-lane half
      float mnew = fmaxf(mrow[r], bm);
      fac[r] = __expf(mrow[r] - mnew);
      mrow[r] = mnew;
      float p0 = __expf(v0 - mnew);
      float p1 = __expf(v1 - mnew);
      float sum = p0 + p1;
      sum += __shfl_xor(sum, 1);
      sum += __shfl_xor(sum, 2);
      sum += __shfl_xor(sum, 4);
      sum += __shfl_xor(sum, 8);
      lrow[r] = lrow[r] * fac[r] + sum;
      z0[r] *= fac[r]; z1[r] *= fac[r]; z2[r] *= fac[r]; z3[r] *= fac[r];
      tile[(r + lh * 8) * 32 + l16]      = (bf16_t)p0;
      tile[(r + lh * 8) * 32 + 16 + l16] = (bf16_t)p1;
    }
    asm volatile("" ::: "memory");  // LDS is in-order per wave; just pin compiler order
    v16bf pa = ld16(tile + l16 * 32 + lh * 16);

    z0 = WMMA_BF16(pa, ld16(vr), z0);
    z1 = WMMA_BF16(pa, ld16(vr + (size_t)16 * 2048), z1);
    z2 = WMMA_BF16(pa, ld16(vr + (size_t)32 * 2048), z2);
    z3 = WMMA_BF16(pa, ld16(vr + (size_t)48 * 2048), z3);
  }

#pragma unroll
  for (int r = 0; r < 8; ++r) {
    int row = q0 + r + lh * 8;
    float inv = 1.0f / lrow[r];
    bf16_t* zr = Z + (size_t)(b * 2048 + row) * 1024 + n * 64 + l16;
    zr[0]  = (bf16_t)(z0[r] * inv);
    zr[16] = (bf16_t)(z1[r] * inv);
    zr[32] = (bf16_t)(z2[r] * inv);
    zr[48] = (bf16_t)(z3[r] * inv);
  }
}

// ---------------- launch ----------------
extern "C" void kernel_launch(void* const* d_in, const int* in_sizes, int n_in,
                              void* d_out, int out_size, void* d_ws, size_t ws_size,
                              hipStream_t stream) {
  (void)in_sizes; (void)n_in; (void)out_size; (void)ws_size;
  const float* x_q  = (const float*)d_in[0];
  const float* x_kv = (const float*)d_in[1];
  /* d_in[2] = mask: causal mask applied analytically */
  const float* W_Q  = (const float*)d_in[3];
  const float* W_K  = (const float*)d_in[4];
  const float* W_V  = (const float*)d_in[5];
  const float* W_O  = (const float*)d_in[6];
  const float* b_Q  = (const float*)d_in[7];
  const float* b_K  = (const float*)d_in[8];
  const float* b_V  = (const float*)d_in[9];
  const float* b_O  = (const float*)d_in[10];

  float* out   = (float*)d_out;                 // (B,SQ,D)
  float* k_out = out + (size_t)8388608;         // (B,SK,N,H)
  float* v_out = out + (size_t)16777216;        // (B,SK,N,H)

  bf16_t* ws     = (bf16_t*)d_ws;
  bf16_t* xq_bf  = ws;
  bf16_t* xkv_bf = xq_bf  + (size_t)8388608;
  bf16_t* wqT    = xkv_bf + (size_t)8388608;
  bf16_t* wkT    = wqT + (size_t)1048576;
  bf16_t* wvT    = wkT + (size_t)1048576;
  bf16_t* woT    = wvT + (size_t)1048576;
  bf16_t* q_bf   = woT + (size_t)1048576;
  bf16_t* k_bf   = q_bf + (size_t)8388608;
  bf16_t* vt_bf  = k_bf + (size_t)8388608;      // transposed V [b][n][h][s]
  bf16_t* z_bf   = vt_bf + (size_t)8388608;

  cvt_bf16<<<8192, 256, 0, stream>>>((const float4*)x_q,  (v4bf*)xq_bf);
  cvt_bf16<<<8192, 256, 0, stream>>>((const float4*)x_kv, (v4bf*)xkv_bf);
  wtrans<true ><<<4096, 256, 0, stream>>>(W_Q, wqT);
  wtrans<true ><<<4096, 256, 0, stream>>>(W_K, wkT);
  wtrans<true ><<<4096, 256, 0, stream>>>(W_V, wvT);
  wtrans<false><<<4096, 256, 0, stream>>>(W_O, woT);

  dim3 gg(64, 8, 1);
  gemm_bf16<2><<<gg, 256, 0, stream>>>(xq_bf,  wqT, b_Q, nullptr, q_bf, nullptr);
  gemm_bf16<3><<<gg, 256, 0, stream>>>(xkv_bf, wkT, b_K, k_out,   k_bf, nullptr);
  gemm_bf16<5><<<gg, 256, 0, stream>>>(xkv_bf, wvT, b_V, v_out,   nullptr, vt_bf);

  flash_attn<<<dim3(16, 16, 4), 256, 0, stream>>>(q_bf, k_bf, vt_bf, z_bf);

  gemm_bf16<1><<<gg, 256, 0, stream>>>(z_bf, woT, b_O, out, nullptr, nullptr);
}
